// MultiHeadAttention_10368051052826
// MI455X (gfx1250) — compile-verified, matrix operand order checked by reading, not run on hardware
//
#include <hip/hip_runtime.h>
#include <hip/hip_bf16.h>

#define D_MODEL 1024
#define NUM_HEADS 16
#define DEPTH 64
#define BATCH 2
#define SEQ 2048

typedef __attribute__((ext_vector_type(16))) __bf16 v16bf;
typedef __attribute__((ext_vector_type(8)))  float  v8f;

union BF16Frag {
    v16bf v;
    uint4 q[2];
    __bf16 h[16];
};

__device__ inline __bf16 f2bf(float f) {
    unsigned u = __builtin_bit_cast(unsigned, f);
    unsigned r = (u + 0x7FFFu + ((u >> 16) & 1u)) >> 16;
    return __builtin_bit_cast(__bf16, (unsigned short)r);
}

// A/B fragment for v_wmma_f32_16x16x32_bf16.
// Lane L holds row/col (L&15); K chunks: {half*8 .. half*8+7} and {16+half*8 ..}.
__device__ inline v16bf load_frag_bf16(const __bf16* rowBase, int half) {
    BF16Frag f;
    const uint4* p = (const uint4*)rowBase;   // 8 bf16 per uint4
    f.q[0] = p[half];
    f.q[1] = p[2 + half];
    return f.v;
}

__device__ inline v8f wmma_bf16(v16bf a, v16bf b, v8f c) {
    return __builtin_amdgcn_wmma_f32_16x16x32_bf16(
        /*neg_a=*/false, a, /*neg_b=*/false, b,
        /*c_mod=*/(short)0, c, /*reuse_a=*/false, /*reuse_b=*/false);
}

#define V8F_ZERO ((v8f){0.f,0.f,0.f,0.f,0.f,0.f,0.f,0.f})

// ---------------- fp32 -> bf16 bulk convert (activations) ----------------
__global__ __launch_bounds__(256) void cvt_bf16_kernel(
    const float* __restrict__ src, __bf16* __restrict__ dst, int n8)
{
    int i = blockIdx.x * 256 + threadIdx.x;
    if (i >= n8) return;
    const float4* p = (const float4*)src + (size_t)i * 2;
    float4 a = p[0], b = p[1];
    BF16Frag f;
    f.h[0] = f2bf(a.x); f.h[1] = f2bf(a.y); f.h[2] = f2bf(a.z); f.h[3] = f2bf(a.w);
    f.h[4] = f2bf(b.x); f.h[5] = f2bf(b.y); f.h[6] = f2bf(b.z); f.h[7] = f2bf(b.w);
    ((uint4*)dst)[i] = f.q[0];
}

// ---------------- weight transpose + bf16 convert: WT[j][k] = W[k][j] ----------------
__global__ __launch_bounds__(256) void transpose_w_kernel(
    const float* __restrict__ W0, const float* __restrict__ W1,
    const float* __restrict__ W2, const float* __restrict__ W3,
    __bf16* __restrict__ T0, __bf16* __restrict__ T1,
    __bf16* __restrict__ T2, __bf16* __restrict__ T3)
{
    const float* W = (blockIdx.z == 0) ? W0 : (blockIdx.z == 1) ? W1 : (blockIdx.z == 2) ? W2 : W3;
    __bf16* T = (blockIdx.z == 0) ? T0 : (blockIdx.z == 1) ? T1 : (blockIdx.z == 2) ? T2 : T3;
    __shared__ float tile[16][17];
    int tx = threadIdx.x & 15, ty = threadIdx.x >> 4;
    tile[ty][tx] = W[(size_t)(blockIdx.y * 16 + ty) * D_MODEL + blockIdx.x * 16 + tx];
    __syncthreads();
    T[(size_t)(blockIdx.x * 16 + ty) * D_MODEL + blockIdx.y * 16 + tx] = f2bf(tile[tx][ty]);
}

// ---------------- projections: P = Xb @ W + b (fully unrolled k-loop) ----------------
// which=0: Q -> Qh (B,H,S,64); which=1: K -> Kh (B,H,S,64); which=2: V -> VT (B,H,64,S)
__global__ __launch_bounds__(256) void proj_kernel(
    const __bf16* __restrict__ Xq, const __bf16* __restrict__ Xk, const __bf16* __restrict__ Xv,
    const __bf16* __restrict__ WTq, const __bf16* __restrict__ WTk, const __bf16* __restrict__ WTv,
    const float* __restrict__ Bq, const float* __restrict__ Bk, const float* __restrict__ Bv,
    __bf16* __restrict__ Qh, __bf16* __restrict__ Kh, __bf16* __restrict__ VT)
{
    int which = blockIdx.y;
    const __bf16* X    = (which == 0) ? Xq  : (which == 1) ? Xk  : Xv;
    const __bf16* WT   = (which == 0) ? WTq : (which == 1) ? WTk : WTv;
    const float*  bias = (which == 0) ? Bq  : (which == 1) ? Bk  : Bv;

    int wid = threadIdx.x >> 5, lane = threadIdx.x & 31;
    int half = lane >> 4, lo = lane & 15;
    int tile = blockIdx.x * 8 + wid;  // 4096 wave-tiles: 256 M-tiles x 16 N-tiles
    int tm = tile >> 4, tn = tile & 15;

    const __bf16* arow = X + (size_t)(tm * 16 + lo) * D_MODEL;
    const __bf16* brow[4];
#pragma unroll
    for (int nt = 0; nt < 4; ++nt)
        brow[nt] = WT + (size_t)(tn * 64 + nt * 16 + lo) * D_MODEL;

    v8f acc[4] = {V8F_ZERO, V8F_ZERO, V8F_ZERO, V8F_ZERO};

#pragma unroll
    for (int kt = 0; kt < D_MODEL / 32; ++kt) {
        v16bf a = load_frag_bf16(arow + kt * 32, half);
        v16bf b0 = load_frag_bf16(brow[0] + kt * 32, half);
        v16bf b1 = load_frag_bf16(brow[1] + kt * 32, half);
        v16bf b2 = load_frag_bf16(brow[2] + kt * 32, half);
        v16bf b3 = load_frag_bf16(brow[3] + kt * 32, half);
        acc[0] = wmma_bf16(a, b0, acc[0]);
        acc[1] = wmma_bf16(a, b1, acc[1]);
        acc[2] = wmma_bf16(a, b2, acc[2]);
        acc[3] = wmma_bf16(a, b3, acc[3]);
    }

#pragma unroll
    for (int nt = 0; nt < 4; ++nt) {
        int j = tn * 64 + nt * 16 + lo;
        float bj = bias[j];
        int h = j >> 6, d = j & 63;
#pragma unroll
        for (int r = 0; r < 8; ++r) {
            int i = tm * 16 + r + half * 8;   // token index over B*S
            int b = i >> 11, s = i & (SEQ - 1);
            float val = acc[nt][r] + bj;
            if (which == 2)
                VT[(((size_t)(b * NUM_HEADS + h)) * DEPTH + d) * SEQ + s] = f2bf(val);
            else if (which == 0)
                Qh[(((size_t)(b * NUM_HEADS + h)) * SEQ + s) * DEPTH + d] = f2bf(val);
            else
                Kh[(((size_t)(b * NUM_HEADS + h)) * SEQ + s) * DEPTH + d] = f2bf(val);
        }
    }
}

// ---------------- logits = Q K^T * scale + mask * -1e9 (fp32, into d_out attn region) ----------------
__global__ __launch_bounds__(256) void logits_kernel(
    const __bf16* __restrict__ Qh, const __bf16* __restrict__ Kh,
    const float* __restrict__ mask, float* __restrict__ logits)
{
    int wid = threadIdx.x >> 5, lane = threadIdx.x & 31;
    int half = lane >> 4, lo = lane & 15;
    int t = blockIdx.x * 8 + wid;    // 4096 waves: (b*H+h) * 128 qtiles
    int bh = t >> 7, qt = t & 127;
    int b = bh >> 4;

    const __bf16* qrow = Qh + (((size_t)bh) * SEQ + qt * 16 + lo) * DEPTH;
    v16bf a0 = load_frag_bf16(qrow, half);
    v16bf a1 = load_frag_bf16(qrow + 32, half);

    const float scale = 0.125f;  // 1/sqrt(64)
    float* outBase = logits + (((size_t)bh) * SEQ + qt * 16) * SEQ;
    const __bf16* kbase = Kh + ((size_t)bh) * SEQ * DEPTH + (size_t)lo * DEPTH;
    const float* mbase = mask + b * SEQ + lo;

    // ping-pong K-tile fragments: set e = even kt, set o = odd kt
    v16bf be0 = load_frag_bf16(kbase, half);
    v16bf be1 = load_frag_bf16(kbase + 32, half);
    float mke = mbase[0] * -1e9f;
    v16bf bo0, bo1;
    float mko;

    const int KT = SEQ / 16;   // 128, even
    for (int kt = 0; kt < KT; kt += 2) {
        int k1 = kt + 1;
        int k2 = (kt + 2 < KT) ? kt + 2 : kt;

        const __bf16* kr1 = kbase + (size_t)k1 * 16 * DEPTH;
        bo0 = load_frag_bf16(kr1, half);
        bo1 = load_frag_bf16(kr1 + 32, half);
        mko = mbase[k1 * 16] * -1e9f;
        {
            v8f acc = V8F_ZERO;
            acc = wmma_bf16(a0, be0, acc);
            acc = wmma_bf16(a1, be1, acc);
#pragma unroll
            for (int r = 0; r < 8; ++r)
                outBase[(size_t)(r + half * 8) * SEQ + kt * 16 + lo] = acc[r] * scale + mke;
        }

        const __bf16* kr2 = kbase + (size_t)k2 * 16 * DEPTH;
        be0 = load_frag_bf16(kr2, half);
        be1 = load_frag_bf16(kr2 + 32, half);
        mke = mbase[k2 * 16] * -1e9f;
        {
            v8f acc = V8F_ZERO;
            acc = wmma_bf16(a0, bo0, acc);
            acc = wmma_bf16(a1, bo1, acc);
#pragma unroll
            for (int r = 0; r < 8; ++r)
                outBase[(size_t)(r + half * 8) * SEQ + k1 * 16 + lo] = acc[r] * scale + mko;
        }
    }
}

// ---------------- row softmax (axis=-1) + P @ V -> concat (B,S,D) bf16 ----------------
__global__ __launch_bounds__(256) void softmax_pv_kernel(
    const float* __restrict__ logits, const __bf16* __restrict__ VT,
    __bf16* __restrict__ concat)
{
    int wid = threadIdx.x >> 5, lane = threadIdx.x & 31;
    int half = lane >> 4, lo = lane & 15;
    int t = blockIdx.x * 8 + wid;
    int bh = t >> 7, qt = t & 127;
    int b = bh >> 4, h = bh & 15;

    const float* row = logits + (((size_t)bh) * SEQ + qt * 16 + lo) * SEQ;
    const float* rh = row + half * (SEQ / 2);

    // pass 1: max; lanes L and L^16 handle the same row, halves of columns
    float m = -3.4e38f;
    for (int c = 0; c < SEQ / 2; c += 4) {
        float4 x = *(const float4*)(rh + c);
        m = fmaxf(m, fmaxf(fmaxf(x.x, x.y), fmaxf(x.z, x.w)));
    }
    m = fmaxf(m, __shfl_xor(m, 16));
    // pass 2: sum of exp
    float s = 0.f;
    for (int c = 0; c < SEQ / 2; c += 4) {
        float4 x = *(const float4*)(rh + c);
        s += __expf(x.x - m) + __expf(x.y - m) + __expf(x.z - m) + __expf(x.w - m);
    }
    s += __shfl_xor(s, 16);
    float inv = 1.0f / s;

    v8f acc[4] = {V8F_ZERO, V8F_ZERO, V8F_ZERO, V8F_ZERO};
    const __bf16* vbase = VT + ((size_t)bh) * DEPTH * SEQ + (size_t)lo * SEQ;

    for (int kt = 0; kt < SEQ / 32; ++kt) {
        int c0 = kt * 32 + half * 8;
        float4 x0 = *(const float4*)(row + c0);
        float4 x1 = *(const float4*)(row + c0 + 4);
        float4 y0 = *(const float4*)(row + c0 + 16);
        float4 y1 = *(const float4*)(row + c0 + 20);
        BF16Frag a;
        a.h[0]  = f2bf(__expf(x0.x - m) * inv); a.h[1]  = f2bf(__expf(x0.y - m) * inv);
        a.h[2]  = f2bf(__expf(x0.z - m) * inv); a.h[3]  = f2bf(__expf(x0.w - m) * inv);
        a.h[4]  = f2bf(__expf(x1.x - m) * inv); a.h[5]  = f2bf(__expf(x1.y - m) * inv);
        a.h[6]  = f2bf(__expf(x1.z - m) * inv); a.h[7]  = f2bf(__expf(x1.w - m) * inv);
        a.h[8]  = f2bf(__expf(y0.x - m) * inv); a.h[9]  = f2bf(__expf(y0.y - m) * inv);
        a.h[10] = f2bf(__expf(y0.z - m) * inv); a.h[11] = f2bf(__expf(y0.w - m) * inv);
        a.h[12] = f2bf(__expf(y1.x - m) * inv); a.h[13] = f2bf(__expf(y1.y - m) * inv);
        a.h[14] = f2bf(__expf(y1.z - m) * inv); a.h[15] = f2bf(__expf(y1.w - m) * inv);
#pragma unroll
        for (int nt = 0; nt < 4; ++nt) {
            v16bf bf = load_frag_bf16(vbase + (size_t)(nt * 16) * SEQ + kt * 32, half);
            acc[nt] = wmma_bf16(a.v, bf, acc[nt]);
        }
    }
#pragma unroll
    for (int nt = 0; nt < 4; ++nt) {
#pragma unroll
        for (int r = 0; r < 8; ++r) {
            int q = qt * 16 + r + half * 8;
            concat[(((size_t)b) * SEQ + q) * D_MODEL + h * DEPTH + nt * 16 + lo] =
                f2bf(acc[nt][r]);
        }
    }
}

// ---------------- in-place softmax over query axis (axis=2) ----------------
__global__ __launch_bounds__(256) void colsoftmax_kernel(float* __restrict__ logits)
{
    int bh = blockIdx.y;
    int c = blockIdx.x * 256 + threadIdx.x;
    float* base = logits + ((size_t)bh) * SEQ * SEQ + c;
    float m = -3.4e38f;
    for (int q = 0; q < SEQ; ++q) m = fmaxf(m, base[(size_t)q * SEQ]);
    float s = 0.f;
    for (int q = 0; q < SEQ; ++q) s += __expf(base[(size_t)q * SEQ] - m);
    float inv = 1.0f / s;
    for (int q = 0; q < SEQ; ++q) {
        size_t idx = (size_t)q * SEQ;
        base[idx] = __expf(base[idx] - m) * inv;
    }
}

// ---------------- output = concat @ Wo + b (fp32 out, fully unrolled k-loop) ----------------
__global__ __launch_bounds__(256) void out_gemm_kernel(
    const __bf16* __restrict__ concat, const __bf16* __restrict__ WTo,
    const float* __restrict__ bias, float* __restrict__ out)
{
    int wid = threadIdx.x >> 5, lane = threadIdx.x & 31;
    int half = lane >> 4, lo = lane & 15;
    int tile = blockIdx.x * 8 + wid;
    int tm = tile >> 4, tn = tile & 15;

    const __bf16* arow = concat + (size_t)(tm * 16 + lo) * D_MODEL;
    const __bf16* brow[4];
#pragma unroll
    for (int nt = 0; nt < 4; ++nt)
        brow[nt] = WTo + (size_t)(tn * 64 + nt * 16 + lo) * D_MODEL;

    v8f acc[4] = {V8F_ZERO, V8F_ZERO, V8F_ZERO, V8F_ZERO};

#pragma unroll
    for (int kt = 0; kt < D_MODEL / 32; ++kt) {
        v16bf a = load_frag_bf16(arow + kt * 32, half);
        v16bf b0 = load_frag_bf16(brow[0] + kt * 32, half);
        v16bf b1 = load_frag_bf16(brow[1] + kt * 32, half);
        v16bf b2 = load_frag_bf16(brow[2] + kt * 32, half);
        v16bf b3 = load_frag_bf16(brow[3] + kt * 32, half);
        acc[0] = wmma_bf16(a, b0, acc[0]);
        acc[1] = wmma_bf16(a, b1, acc[1]);
        acc[2] = wmma_bf16(a, b2, acc[2]);
        acc[3] = wmma_bf16(a, b3, acc[3]);
    }

#pragma unroll
    for (int nt = 0; nt < 4; ++nt) {
        int j = tn * 64 + nt * 16 + lo;
        float bj = bias[j];
#pragma unroll
        for (int r = 0; r < 8; ++r) {
            int i = tm * 16 + r + half * 8;
            out[(size_t)i * D_MODEL + j] = acc[nt][r] + bj;
        }
    }
}

extern "C" void kernel_launch(void* const* d_in, const int* in_sizes, int n_in,
                              void* d_out, int out_size, void* d_ws, size_t ws_size,
                              hipStream_t stream) {
    (void)in_sizes; (void)n_in; (void)out_size; (void)ws_size;

    // inputs: v, k, q, mask, training, wq_w, wq_b, wk_w, wk_b, wv_w, wv_b, wo_w, wo_b
    const float* v    = (const float*)d_in[0];
    const float* k    = (const float*)d_in[1];
    const float* q    = (const float*)d_in[2];
    const float* mask = (const float*)d_in[3];
    const float* wq_w = (const float*)d_in[5];
    const float* wq_b = (const float*)d_in[6];
    const float* wk_w = (const float*)d_in[7];
    const float* wk_b = (const float*)d_in[8];
    const float* wv_w = (const float*)d_in[9];
    const float* wv_b = (const float*)d_in[10];
    const float* wo_w = (const float*)d_in[11];
    const float* wo_b = (const float*)d_in[12];

    char* ws = (char*)d_ws;
    const size_t MB = 1024 * 1024;
    __bf16* WTq    = (__bf16*)(ws + 0 * MB);
    __bf16* WTk    = (__bf16*)(ws + 2 * MB);
    __bf16* WTv    = (__bf16*)(ws + 4 * MB);
    __bf16* WTo    = (__bf16*)(ws + 6 * MB);
    __bf16* Qh     = (__bf16*)(ws + 8 * MB);   // (B,H,S,64)
    __bf16* Kh     = (__bf16*)(ws + 16 * MB);  // (B,H,S,64)
    __bf16* VT     = (__bf16*)(ws + 24 * MB);  // (B,H,64,S)
    __bf16* concat = (__bf16*)(ws + 32 * MB);  // (B,S,D_MODEL)
    __bf16* Xqb    = (__bf16*)(ws + 40 * MB);  // (B*S, D_MODEL) bf16
    __bf16* Xkb    = (__bf16*)(ws + 48 * MB);
    __bf16* Xvb    = (__bf16*)(ws + 56 * MB);

    float* out_main = (float*)d_out;                                  // (B,S,D)
    float* logits   = out_main + (size_t)BATCH * SEQ * D_MODEL;       // (B,H,S,S)

    const int n8 = BATCH * SEQ * D_MODEL / 8;
    const int cvt_blocks = (n8 + 255) / 256;

    transpose_w_kernel<<<dim3(64, 64, 4), 256, 0, stream>>>(
        wq_w, wk_w, wv_w, wo_w, WTq, WTk, WTv, WTo);

    cvt_bf16_kernel<<<cvt_blocks, 256, 0, stream>>>(q, Xqb, n8);
    cvt_bf16_kernel<<<cvt_blocks, 256, 0, stream>>>(k, Xkb, n8);
    cvt_bf16_kernel<<<cvt_blocks, 256, 0, stream>>>(v, Xvb, n8);

    proj_kernel<<<dim3(512, 3), 256, 0, stream>>>(
        Xqb, Xkb, Xvb, WTq, WTk, WTv, wq_b, wk_b, wv_b, Qh, Kh, VT);

    logits_kernel<<<512, 256, 0, stream>>>(Qh, Kh, mask, logits);

    softmax_pv_kernel<<<512, 256, 0, stream>>>(logits, VT, concat);

    colsoftmax_kernel<<<dim3(8, 32), 256, 0, stream>>>(logits);

    out_gemm_kernel<<<512, 256, 0, stream>>>(concat, WTo, wo_b, out_main);
}
